// Attention_9156870275268
// MI455X (gfx1250) — matrix operand order repaired and checked by reading, not compiled, not run on hardware
//
#include <hip/hip_runtime.h>
#include <hip/hip_bf16.h>
#include <math.h>

typedef __attribute__((ext_vector_type(16))) __bf16 v16bf;
typedef __attribute__((ext_vector_type(8)))  float  v8f;

#define B_      64
#define S_      1024
#define D_      1024          // QD = KD = VD = AD
#define KSTRIDE 1032          // 1024 + 8 ushort pad -> kills LDS bank conflicts

union AFrag { v16bf v; uint4 q[2]; };

static __device__ __forceinline__ unsigned short f2bf(float f) {
  unsigned u = __builtin_bit_cast(unsigned, f);
  u += 0x7FFFu + ((u >> 16) & 1u);          // round-to-nearest-even
  return (unsigned short)(u >> 16);
}
static __device__ __forceinline__ unsigned pack2(float lo, float hi) {
  return (unsigned)f2bf(lo) | ((unsigned)f2bf(hi) << 16);
}

// Hardware tanh (CDNA5 V_TANH_F32 trans op). v_nop covers the 1-op TRANS
// result hazard since the compiler can't schedule around raw inline asm.
static __device__ __forceinline__ float htanh(float x) {
  float r;
  asm volatile("v_tanh_f32 %0, %1\n\tv_nop" : "=v"(r) : "v"(x));
  return r;
}

// ---------------- Pass 0: Wk (f32 [A,K]) -> bf16 ----------------
__global__ __launch_bounds__(256)
void cvt_wk_kernel(const float* __restrict__ wk, unsigned short* __restrict__ wkbf) {
  size_t i = ((size_t)blockIdx.x * blockDim.x + threadIdx.x) * 4;
  float4 f = *(const float4*)(wk + i);
  uint2 u; u.x = pack2(f.x, f.y); u.y = pack2(f.z, f.w);
  *(uint2*)(wkbf + i) = u;
}

// ---------------- Pass 1: pqp[b,a] = query[b]·Wq[a] + bq[a] + bk[a] ----------------
__global__ __launch_bounds__(256)
void pq_kernel(const float* __restrict__ query, const float* __restrict__ Wq,
               const float* __restrict__ bq, const float* __restrict__ bk,
               float* __restrict__ pqp) {
  __shared__ float qs[D_];
  const int b = blockIdx.y;
  const int a = blockIdx.x * 256 + threadIdx.x;
  {
    int k = threadIdx.x * 4;
    *(float4*)(qs + k) = *(const float4*)(query + (size_t)b * D_ + k);
  }
  __syncthreads();
  const float* wr = Wq + (size_t)a * D_;
  float acc = 0.0f;
  for (int k = 0; k < D_; k += 4) {
    float4 w = *(const float4*)(wr + k);
    acc += qs[k] * w.x + qs[k + 1] * w.y + qs[k + 2] * w.z + qs[k + 3] * w.w;
  }
  pqp[(size_t)b * D_ + a] = acc + bq[a] + bk[a];
}

// ---------------- Pass 2: fused WMMA scores kernel ----------------
// WG = (b, 64 s-rows). keys tile staged fp32->bf16 in dynamic LDS (132 KB).
// Wave w covers a-tiles [8w..8w+7]; register blocking 4 s-subtiles x 2 a-tiles.
__global__ __launch_bounds__(256)
void scores_kernel(const float* __restrict__ keys,
                   const unsigned short* __restrict__ wkbf,
                   const float* __restrict__ pqp,
                   const float* __restrict__ Ws,
                   float* __restrict__ scores) {
  extern __shared__ char smem[];
  unsigned short* ktile = (unsigned short*)smem;                          // [64][KSTRIDE]
  float* sscore = (float*)(smem + 64 * KSTRIDE * sizeof(unsigned short)); // [64]

  const int b   = blockIdx.y;
  const int s0  = blockIdx.x * 64;
  const int tid = threadIdx.x;

  if (tid < 64) sscore[tid] = 0.0f;

  // Stage keys[b, s0:s0+64, :] as bf16 into LDS (4 threads per row).
  {
    const int r  = tid >> 2;
    const int c0 = (tid & 3) * 256;
    const float* src = keys + ((size_t)b * S_ + s0 + r) * D_ + c0;
    unsigned short* dst = ktile + r * KSTRIDE + c0;
    #pragma unroll 4
    for (int c = 0; c < 256; c += 4) {
      float4 f = *(const float4*)(src + c);
      uint2 u; u.x = pack2(f.x, f.y); u.y = pack2(f.z, f.w);
      *(uint2*)(dst + c) = u;
    }
  }
  __syncthreads();

  const int w    = tid >> 5;
  const int lane = tid & 31;
  const int lm   = lane & 15;
  const int half = lane >> 4;

  for (int it = 0; it < 4; ++it) {
    const int aBase = (w * 8 + it * 2) * 16;   // 2 a-tiles: aBase, aBase+16
    v8f acc[8] = {};                            // [sg*2 + j]

    #pragma unroll 1
    for (int k0 = 0; k0 < D_; k0 += 32) {
      AFrag a[4];
      #pragma unroll
      for (int sg = 0; sg < 4; ++sg) {
        const unsigned short* rp = ktile + (sg * 16 + lm) * KSTRIDE + k0;
        a[sg].q[0] = *(const uint4*)(rp + 8 * half);
        a[sg].q[1] = *(const uint4*)(rp + 16 + 8 * half);
      }
      AFrag bm[2];
      #pragma unroll
      for (int j = 0; j < 2; ++j) {
        const unsigned short* wp =
            wkbf + (size_t)(aBase + 16 * j + lm) * D_ + k0 + 8 * half;
        bm[j].q[0] = *(const uint4*)(wp);
        bm[j].q[1] = *(const uint4*)(wp + 16);
      }
      #pragma unroll
      for (int sg = 0; sg < 4; ++sg)
        #pragma unroll
        for (int j = 0; j < 2; ++j)
          acc[sg * 2 + j] = __builtin_amdgcn_wmma_f32_16x16x32_bf16(
              false, a[sg].v, false, bm[j].v, (short)0, acc[sg * 2 + j],
              false, false);
    }

    // Epilogue for this a-pair: tanh (HW trans op) + Ws dot, fuse j=0/1,
    // reduce across the 16 lm-lanes, one LDS atomic per (wave,row).
    const float p0 = pqp[(size_t)b * D_ + aBase + lm];
    const float p1 = pqp[(size_t)b * D_ + aBase + 16 + lm];
    const float w0 = Ws[aBase + lm];
    const float w1 = Ws[aBase + 16 + lm];
    #pragma unroll
    for (int sg = 0; sg < 4; ++sg)
      #pragma unroll
      for (int i = 0; i < 8; ++i) {
        float v = htanh(acc[sg * 2 + 0][i] + p0) * w0
                + htanh(acc[sg * 2 + 1][i] + p1) * w1;
        v += __shfl_xor(v, 1, 32);
        v += __shfl_xor(v, 2, 32);
        v += __shfl_xor(v, 4, 32);
        v += __shfl_xor(v, 8, 32);
        if (lm == 0) atomicAdd(&sscore[sg * 16 + i + 8 * half], v);
      }
  }

  __syncthreads();
  if (tid < 64) scores[(size_t)b * S_ + s0 + tid] = sscore[tid];
}

// ---------------- Pass 3: softmax over S; writes ws attn + d_out attn ----------------
__global__ __launch_bounds__(256)
void softmax_kernel(const float* __restrict__ scores,
                    float* __restrict__ attn, float* __restrict__ out_attn) {
  __shared__ float red[256];
  const int b = blockIdx.x, tid = threadIdx.x;
  const float* sr = scores + (size_t)b * S_;
  float4 v = *(const float4*)(sr + tid * 4);
  float m = fmaxf(fmaxf(v.x, v.y), fmaxf(v.z, v.w));
  red[tid] = m; __syncthreads();
  for (int off = 128; off > 0; off >>= 1) {
    if (tid < off) red[tid] = fmaxf(red[tid], red[tid + off]);
    __syncthreads();
  }
  const float bm = red[0]; __syncthreads();
  float4 e;
  e.x = expf(v.x - bm); e.y = expf(v.y - bm);
  e.z = expf(v.z - bm); e.w = expf(v.w - bm);
  red[tid] = e.x + e.y + e.z + e.w; __syncthreads();
  for (int off = 128; off > 0; off >>= 1) {
    if (tid < off) red[tid] += red[tid + off];
    __syncthreads();
  }
  const float inv = 1.0f / red[0];
  float4 o; o.x = e.x * inv; o.y = e.y * inv; o.z = e.z * inv; o.w = e.w * inv;
  *(float4*)(attn + (size_t)b * S_ + tid * 4) = o;
  *(float4*)(out_attn + (size_t)b * S_ + tid * 4) = o;
}

// ---------------- Pass 4a: partial context over 64-row s-chunks ----------------
__global__ __launch_bounds__(256)
void ctx_partial_kernel(const float* __restrict__ attn,
                        const float* __restrict__ values,
                        float* __restrict__ partial) {
  const int cx = blockIdx.x, b = blockIdx.y, tid = threadIdx.x;
  const int v0 = tid * 4;
  const int sBase = cx * 64;
  const float* ar = attn + (size_t)b * S_;
  float ax = 0.f, ay = 0.f, az = 0.f, aw = 0.f;
  for (int s = 0; s < 64; ++s) {
    const float wgt = ar[sBase + s];
    float4 val = *(const float4*)(values + ((size_t)b * S_ + sBase + s) * D_ + v0);
    ax += wgt * val.x; ay += wgt * val.y; az += wgt * val.z; aw += wgt * val.w;
  }
  float4 o; o.x = ax; o.y = ay; o.z = az; o.w = aw;
  *(float4*)(partial + ((size_t)cx * B_ + b) * D_ + v0) = o;
}

// ---------------- Pass 4b: reduce 16 partials -> context ----------------
__global__ __launch_bounds__(256)
void ctx_reduce_kernel(const float* __restrict__ partial, float* __restrict__ outctx) {
  const int b = blockIdx.x, tid = threadIdx.x;
  const int v0 = tid * 4;
  float ax = 0.f, ay = 0.f, az = 0.f, aw = 0.f;
  for (int cx = 0; cx < 16; ++cx) {
    float4 p = *(const float4*)(partial + ((size_t)cx * B_ + b) * D_ + v0);
    ax += p.x; ay += p.y; az += p.z; aw += p.w;
  }
  float4 o; o.x = ax; o.y = ay; o.z = az; o.w = aw;
  *(float4*)(outctx + (size_t)b * D_ + v0) = o;
}

extern "C" void kernel_launch(void* const* d_in, const int* in_sizes, int n_in,
                              void* d_out, int out_size, void* d_ws, size_t ws_size,
                              hipStream_t stream) {
  const float* query = (const float*)d_in[0];   // [64,1024]
  const float* keys  = (const float*)d_in[1];   // [64,1024,1024]
  const float* vals  = (const float*)d_in[2];   // [64,1024,1024]
  const float* Wq    = (const float*)d_in[3];   // [1024,1024]
  const float* bq    = (const float*)d_in[4];   // [1024]
  const float* Wk    = (const float*)d_in[5];   // [1024,1024]
  const float* bk    = (const float*)d_in[6];   // [1024]
  const float* Ws    = (const float*)d_in[7];   // [1024]

  float* out_ctx  = (float*)d_out;                   // [64,1024]
  float* out_attn = (float*)d_out + (size_t)B_ * S_; // [64,1024]

  // Workspace layout
  unsigned short* wkbf = (unsigned short*)d_ws;                         // 2 MB
  float* pqp     = (float*)((char*)d_ws + (size_t)2 * 1024 * 1024);     // 256 KB
  float* scores  = pqp + (size_t)B_ * D_;                               // 256 KB
  float* attn    = scores + (size_t)B_ * S_;                            // 256 KB
  float* partial = attn + (size_t)B_ * S_;                              // 4 MB

  cvt_wk_kernel<<<1024, 256, 0, stream>>>(Wk, wkbf);
  pq_kernel<<<dim3(4, B_), 256, 0, stream>>>(query, Wq, bq, bk, pqp);

  const size_t smem = (size_t)64 * KSTRIDE * sizeof(unsigned short) + 64 * sizeof(float);
  scores_kernel<<<dim3(S_ / 64, B_), 256, smem, stream>>>(keys, wkbf, pqp, Ws, scores);

  softmax_kernel<<<B_, 256, 0, stream>>>(scores, attn, out_attn);
  ctx_partial_kernel<<<dim3(16, B_), 256, 0, stream>>>(attn, vals, partial);
  ctx_reduce_kernel<<<B_, 256, 0, stream>>>(partial, out_ctx);
}
